// Qwen3_5Experts_85529978732979
// MI455X (gfx1250) — compile-verified, matrix operand order checked by reading, not run on hardware
//
#include <hip/hip_runtime.h>
#include <stdint.h>

#define E_    32
#define TOPK_ 8
#define H_    2048
#define I_    1024
#define AMAX_ 2
#define R_    16
#define T_    4096
#define TE_   (T_ * TOPK_)     // 32768 expanded tokens
#define MAXMT 320              // >= worst-case sum(ceil(cnt/128)) = 288
#define LDSP  40               // padded LDS row stride in halves (32 data + 8 pad)

typedef __attribute__((ext_vector_type(16))) __bf16 v16bf;
typedef __attribute__((ext_vector_type(8)))  __bf16 v8bf;
typedef __attribute__((ext_vector_type(8)))  float  v8f;
typedef __attribute__((ext_vector_type(4)))  unsigned int u32x4;
typedef __attribute__((ext_vector_type(8)))  int    i32x8;
typedef __attribute__((ext_vector_type(4)))  int    i32x4;

__device__ __forceinline__ __bf16 tobf(float f) { return (__bf16)f; }

// ---------------------------------------------------------------------------
// Tensor Data Mover: async 2D tile load Global->LDS.
// tile_dim0 = 32 halves along K (contiguous), tile_dim1 = rows (strided).
// LDS padding: every 16 DWORDs insert 4 -> LDS row stride 40 halves (80B,
// 16B aligned so fragment reads are ds_load_b128 with conflict-free banking).
// ---------------------------------------------------------------------------
__device__ __forceinline__ void tdm_load_tile(uint64_t gaddr, uint32_t lds_off,
                                              uint32_t rows,
                                              uint32_t row_stride_elems,
                                              uint32_t tensor_rows)
{
  // ---- D# group 0 (128b) ----
  uint32_t d0w0 = 1u;                                  // count=1, user descriptor
  uint32_t d0w1 = lds_off;                             // lds_addr (bytes)
  uint32_t d0w2 = (uint32_t)gaddr;                     // global_addr[31:0]
  uint32_t d0w3 = ((uint32_t)(gaddr >> 32) & 0x01FFFFFFu) | 0x80000000u; // addr[56:32] | type=2
  // ---- D# group 1 (256b) ----
  uint32_t w0 = (1u << 16)    // data_size = 2 bytes
              | (1u << 20)    // pad_enable
              | (3u << 22)    // pad_interval code 3 = 16 DWORDs
              | (3u << 25);   // pad_amount   code 3 = 4 DWORDs
  uint32_t w1 = (32u << 16);                           // tensor_dim0[15:0] = 32
  uint32_t w2 = (tensor_rows & 0xFFFFu) << 16;         // dim0 hi=0 | tensor_dim1[15:0]
  uint32_t w3 = ((tensor_rows >> 16) & 0xFFFFu)        // tensor_dim1[31:16]
              | (32u << 16);                           // tile_dim0 = 32
  uint32_t w4 = rows & 0xFFFFu;                        // tile_dim1 (tile_dim2 = 0)
  uint32_t w5 = row_stride_elems;                      // tensor_dim0_stride[31:0]
  uint32_t w6 = 0u;
  uint32_t w7 = 0u;

  u32x4 g0;
  g0[0] = (unsigned)__builtin_amdgcn_readfirstlane((int)d0w0);
  g0[1] = (unsigned)__builtin_amdgcn_readfirstlane((int)d0w1);
  g0[2] = (unsigned)__builtin_amdgcn_readfirstlane((int)d0w2);
  g0[3] = (unsigned)__builtin_amdgcn_readfirstlane((int)d0w3);
  i32x8 g1;
  g1[0] = __builtin_amdgcn_readfirstlane((int)w0);
  g1[1] = __builtin_amdgcn_readfirstlane((int)w1);
  g1[2] = __builtin_amdgcn_readfirstlane((int)w2);
  g1[3] = __builtin_amdgcn_readfirstlane((int)w3);
  g1[4] = __builtin_amdgcn_readfirstlane((int)w4);
  g1[5] = __builtin_amdgcn_readfirstlane((int)w5);
  g1[6] = __builtin_amdgcn_readfirstlane((int)w6);
  g1[7] = __builtin_amdgcn_readfirstlane((int)w7);
  i32x4 z4 = {0, 0, 0, 0};
#if defined(__clang_major__) && (__clang_major__ >= 23)
  i32x8 z8 = {0, 0, 0, 0, 0, 0, 0, 0};
  __builtin_amdgcn_tensor_load_to_lds(g0, g1, z4, z4, z8, 0);
#else
  __builtin_amdgcn_tensor_load_to_lds(g0, g1, z4, z4, 0);
#endif
}

__device__ __forceinline__ uint32_t lds_addr_of(const void* p) {
  return (uint32_t)(uintptr_t)p;  // low 32 bits of generic address = LDS offset
}

// One wave's 16x32 bf16 fragment from a padded LDS tile (A-operand layout):
// lane 0-15:  row = lane,    K 0-7 & 16-23
// lane 16-31: row = lane-16, K 8-15 & 24-31
__device__ __forceinline__ v16bf load_frag(const __bf16* tile, int row, int lane) {
  const int kb = (lane >> 4) << 3;
  const v8bf a = *(const v8bf*)(tile + row * LDSP + kb);
  const v8bf b = *(const v8bf*)(tile + row * LDSP + kb + 16);
  v16bf f;
#pragma unroll
  for (int i = 0; i < 8; ++i) { f[i] = a[i]; f[i + 8] = b[i]; }
  return f;
}

__device__ __forceinline__ v8f wmma_bf16(v16bf a, v16bf b, v8f c) {
  return __builtin_amdgcn_wmma_f32_16x16x32_bf16(false, a, false, b, (short)0, c,
                                                 false, false);
}

// ---------------------------------------------------------------------------
// Small utility kernels
// ---------------------------------------------------------------------------
__global__ void k_zero_f32(float* p, int n) {
  int i = blockIdx.x * 256 + threadIdx.x;
  if (i < n) p[i] = 0.0f;
}

__global__ void k_zero_u32(unsigned int* p, int n) {
  int i = blockIdx.x * 256 + threadIdx.x;
  if (i < n) p[i] = 0u;
}

__global__ void k_hist(const int* __restrict__ sel, unsigned int* __restrict__ counts) {
  int i = blockIdx.x * 256 + threadIdx.x;
  if (i < TE_) atomicAdd(&counts[sel[i]], 1u);
}

__global__ void k_scan_tiles(const unsigned int* __restrict__ counts,
                             unsigned int* __restrict__ offsets,
                             unsigned int* __restrict__ cursors,
                             unsigned int* __restrict__ mt_e,
                             unsigned int* __restrict__ mt_row,
                             unsigned int* __restrict__ mt_rows,
                             unsigned int* __restrict__ n_mtiles) {
  if (threadIdx.x != 0) return;
  unsigned int off = 0, nm = 0;
  for (int e = 0; e < E_; ++e) {
    offsets[e] = off;
    cursors[e] = off;
    unsigned int c = counts[e];
    for (unsigned int r = 0; r < c; r += 128u) {
      mt_e[nm] = (unsigned int)e;
      mt_row[nm] = off + r;
      unsigned int rem = c - r;
      mt_rows[nm] = rem < 128u ? rem : 128u;
      ++nm;
    }
    off += c;
  }
  offsets[E_] = off;
  n_mtiles[0] = nm;
}

__global__ void k_scatter(const int* __restrict__ sel, const int* __restrict__ adp,
                          const float* __restrict__ rw,
                          unsigned int* __restrict__ cursors,
                          unsigned int* __restrict__ sorted_tok,
                          unsigned int* __restrict__ sorted_adp,
                          float* __restrict__ sorted_rw) {
  int i = blockIdx.x * 256 + threadIdx.x;
  if (i >= TE_) return;
  int t = i >> 3;
  int e = sel[i];
  unsigned int pos = atomicAdd(&cursors[e], 1u);
  sorted_tok[pos] = (unsigned int)t;
  sorted_adp[pos] = (unsigned int)adp[t];
  sorted_rw[pos] = rw[i];
}

// Gather expert-sorted token rows, fp32 -> bf16. One thread = 8 elements.
__global__ void k_gather_x(const float* __restrict__ x,
                           const unsigned int* __restrict__ sorted_tok,
                           __bf16* __restrict__ xs) {
  int linear = blockIdx.x * 256 + threadIdx.x;  // TE_ * (H_/8) total
  int s = linear >> 8;                          // H_/8 = 256 chunks per row
  int off = (linear & 255) << 3;
  const float4* p = (const float4*)(x + (size_t)sorted_tok[s] * H_ + off);
  float4 f0 = p[0], f1 = p[1];
  v8bf o;
  o[0] = tobf(f0.x); o[1] = tobf(f0.y); o[2] = tobf(f0.z); o[3] = tobf(f0.w);
  o[4] = tobf(f1.x); o[5] = tobf(f1.y); o[6] = tobf(f1.z); o[7] = tobf(f1.w);
  *(v8bf*)(xs + (size_t)s * H_ + off) = o;
}

// Transpose-convert weights: src[e][K][N] fp32 -> dst[e][N][K] bf16 (N-major,
// K contiguous, so B tiles have the same TDM shape as A tiles).
__global__ __launch_bounds__(256) void k_cvt_t(const float* __restrict__ src,
                                               __bf16* __restrict__ dst,
                                               int K, int N) {
  __shared__ float tile[32][33];
  int tilesN = N >> 5, tilesK = K >> 5;
  int b = blockIdx.x;
  int nt = b % tilesN;
  int kt = (b / tilesN) % tilesK;
  int e  = b / (tilesN * tilesK);
  int k0 = kt << 5, n0 = nt << 5;
  int c = threadIdx.x & 31, r8 = threadIdx.x >> 5;
  const float* s = src + ((size_t)e * K + k0) * N + n0;
#pragma unroll
  for (int i = 0; i < 4; ++i) {
    int r = r8 + i * 8;
    tile[r][c] = s[(size_t)r * N + c];
  }
  __syncthreads();
  __bf16* d = dst + ((size_t)e * N + n0) * K + k0;
#pragma unroll
  for (int i = 0; i < 4; ++i) {
    int r = r8 + i * 8;
    d[(size_t)r * K + c] = tobf(tile[c][r]);
  }
}

// Acat: src[2][E][K][16] -> dst[e][32][K] bf16 (adapters concatenated on N)
__global__ void k_cat_A(const float* __restrict__ src, __bf16* __restrict__ dst, int K) {
  int idx = blockIdx.x * 256 + threadIdx.x;
  int total = E_ * 32 * K;
  if (idx >= total) return;
  int k = idx % K;
  int c = (idx / K) & 31;
  int e = idx / (K * 32);
  dst[idx] = tobf(src[((((size_t)(c >> 4) * E_ + e) * K + k) * R_) + (c & 15)]);
}

// BcatT: src[2][E][16][N] -> dst[e][N][32] bf16 (adapters concatenated on K, N-major)
__global__ void k_cat_B(const float* __restrict__ src, __bf16* __restrict__ dst, int N) {
  int idx = blockIdx.x * 256 + threadIdx.x;
  int total = E_ * N * 32;
  if (idx >= total) return;
  int kk = idx & 31;
  int n  = (idx / 32) % N;
  int e  = idx / (32 * N);
  dst[idx] = tobf(src[(((size_t)(kk >> 4) * E_ + e) * R_ + (kk & 15)) * N + n]);
}

// ---------------------------------------------------------------------------
// Grouped GEMM: z = x_sorted @ Acat[e]  (N = 32), masked by the row's adapter
// half, stored bf16. Tiles fed by TDM, math on v_wmma_f32_16x16x32_bf16.
// ---------------------------------------------------------------------------
__global__ __launch_bounds__(256)
void k_gemm_z(const __bf16* __restrict__ Asrt, int K,
              const __bf16* __restrict__ AcatT,
              const unsigned int* __restrict__ mt_e,
              const unsigned int* __restrict__ mt_row,
              const unsigned int* __restrict__ mt_rows,
              const unsigned int* __restrict__ n_mtiles,
              const unsigned int* __restrict__ sorted_adp,
              __bf16* __restrict__ zcat) {
  if (blockIdx.x >= n_mtiles[0]) return;
  const int e     = (int)mt_e[blockIdx.x];
  const int row0  = (int)mt_row[blockIdx.x];
  const int rowsv = (int)mt_rows[blockIdx.x];
  const int tid = threadIdx.x, lane = tid & 31, w = tid >> 5;

  __shared__ __attribute__((aligned(16))) __bf16 sA[2][128 * LDSP];
  __shared__ __attribute__((aligned(16))) __bf16 sC[2][32 * LDSP];

  const v8f vz = {0.f, 0.f, 0.f, 0.f, 0.f, 0.f, 0.f, 0.f};
  v8f acc[2] = {vz, vz};

  const int KS = K >> 5;
  const uint64_t aBase = (uint64_t)(uintptr_t)Asrt + ((uint64_t)row0 * K) * 2u;
  const uint64_t cBase = (uint64_t)(uintptr_t)AcatT + ((uint64_t)e * 32 * K) * 2u;

  if (tid < 32) {
    tdm_load_tile(aBase, lds_addr_of(&sA[0][0]), 128u, (uint32_t)K, (uint32_t)rowsv);
    tdm_load_tile(cBase, lds_addr_of(&sC[0][0]), 32u, (uint32_t)K, 32u);
  }
  for (int s = 0; s < KS; ++s) {
    const int buf = s & 1;
    if (tid < 32 && (s + 1) < KS) {
      uint64_t ko = (uint64_t)(s + 1) * 64u;  // 32 halves * 2B
      tdm_load_tile(aBase + ko, lds_addr_of(&sA[(s + 1) & 1][0]), 128u, (uint32_t)K,
                    (uint32_t)rowsv);
      tdm_load_tile(cBase + ko, lds_addr_of(&sC[(s + 1) & 1][0]), 32u, (uint32_t)K, 32u);
    }
    if ((s + 1) < KS) __builtin_amdgcn_s_wait_tensorcnt(2);
    else              __builtin_amdgcn_s_wait_tensorcnt(0);
    __syncthreads();
    v16bf af = load_frag(sA[buf], 16 * w + (lane & 15), lane);
#pragma unroll
    for (int n = 0; n < 2; ++n) {
      v16bf bfg = load_frag(sC[buf], 16 * n + (lane & 15), lane);
      acc[n] = wmma_bf16(af, bfg, acc[n]);
    }
    __syncthreads();
  }
  // Mask to the row's adapter half and store bf16.
  const int col_lo = lane & 15;
#pragma unroll
  for (int v = 0; v < 8; ++v) {
    int m = ((lane >> 4) << 3) + v;
    int lr = 16 * w + m;
    if (lr >= rowsv) continue;
    size_t gr = (size_t)(row0 + lr);
    int a = (int)sorted_adp[gr];
#pragma unroll
    for (int n = 0; n < 2; ++n) {
      int col = 16 * n + col_lo;
      float val = ((col >> 4) == a) ? acc[n][v] : 0.0f;
      zcat[gr * 32 + col] = tobf(val);
    }
  }
}

// ---------------------------------------------------------------------------
// Fused gate+up grouped GEMM. Block tile 128(M) x 128(N), 8 waves in a 4x2
// grid: each wave owns 32 rows x 64 cols for BOTH projections
// (16 v8f accumulators). Per K-stage: 20 ds_load_b128 feed 16 WMMAs.
// Epilogue: h = silu(gate + zG@BcatG) * (up + zU@BcatU), stored bf16.
// ---------------------------------------------------------------------------
__global__ __launch_bounds__(256)
void k_gemm_gateup(const __bf16* __restrict__ Asrt,
                   const __bf16* __restrict__ Wgt,
                   const __bf16* __restrict__ Wut,
                   const __bf16* __restrict__ zcatG,
                   const __bf16* __restrict__ zcatU,
                   const __bf16* __restrict__ BcatGt,
                   const __bf16* __restrict__ BcatUt,
                   const unsigned int* __restrict__ mt_e,
                   const unsigned int* __restrict__ mt_row,
                   const unsigned int* __restrict__ mt_rows,
                   const unsigned int* __restrict__ n_mtiles,
                   __bf16* __restrict__ hs) {
  if (blockIdx.x >= n_mtiles[0]) return;
  const int K = H_, Ntot = I_;
  const int e     = (int)mt_e[blockIdx.x];
  const int row0  = (int)mt_row[blockIdx.x];
  const int rowsv = (int)mt_rows[blockIdx.x];
  const int n0    = blockIdx.y * 128;
  const int tid = threadIdx.x, lane = tid & 31;
  const int wm = (tid >> 5) & 3;   // M slot: rows 32*wm..+32
  const int wn = tid >> 7;         // N slot: cols 64*wn..+64

  __shared__ __attribute__((aligned(16))) __bf16 sA[2][128 * LDSP];
  __shared__ __attribute__((aligned(16))) __bf16 sBg[2][128 * LDSP];
  __shared__ __attribute__((aligned(16))) __bf16 sBu[2][128 * LDSP];
  __shared__ __attribute__((aligned(16))) __bf16 sZg[128 * LDSP];
  __shared__ __attribute__((aligned(16))) __bf16 sZu[128 * LDSP];
  __shared__ __attribute__((aligned(16))) __bf16 sCg[128 * LDSP];
  __shared__ __attribute__((aligned(16))) __bf16 sCu[128 * LDSP];

  const v8f vz = {0.f, 0.f, 0.f, 0.f, 0.f, 0.f, 0.f, 0.f};
  v8f accg[2][4], accu[2][4];
#pragma unroll
  for (int i = 0; i < 2; ++i)
#pragma unroll
    for (int j = 0; j < 4; ++j) { accg[i][j] = vz; accu[i][j] = vz; }

  const int KS = K >> 5;
  const uint64_t aBase = (uint64_t)(uintptr_t)Asrt + ((uint64_t)row0 * K) * 2u;
  const uint64_t gBase =
      (uint64_t)(uintptr_t)Wgt + (((uint64_t)e * Ntot + n0) * (uint64_t)K) * 2u;
  const uint64_t uBase =
      (uint64_t)(uintptr_t)Wut + (((uint64_t)e * Ntot + n0) * (uint64_t)K) * 2u;

  if (tid < 32) {
    tdm_load_tile(aBase, lds_addr_of(&sA[0][0]), 128u, (uint32_t)K, (uint32_t)rowsv);
    tdm_load_tile(gBase, lds_addr_of(&sBg[0][0]), 128u, (uint32_t)K, 128u);
    tdm_load_tile(uBase, lds_addr_of(&sBu[0][0]), 128u, (uint32_t)K, 128u);
    tdm_load_tile((uint64_t)(uintptr_t)zcatG + ((uint64_t)row0 * 32u) * 2u,
                  lds_addr_of(&sZg[0]), 128u, 32u, (uint32_t)rowsv);
    tdm_load_tile((uint64_t)(uintptr_t)zcatU + ((uint64_t)row0 * 32u) * 2u,
                  lds_addr_of(&sZu[0]), 128u, 32u, (uint32_t)rowsv);
    tdm_load_tile((uint64_t)(uintptr_t)BcatGt + (((uint64_t)e * Ntot + n0) * 32u) * 2u,
                  lds_addr_of(&sCg[0]), 128u, 32u, 128u);
    tdm_load_tile((uint64_t)(uintptr_t)BcatUt + (((uint64_t)e * Ntot + n0) * 32u) * 2u,
                  lds_addr_of(&sCu[0]), 128u, 32u, 128u);
  }
  for (int s = 0; s < KS; ++s) {
    const int buf = s & 1;
    if (tid < 32 && (s + 1) < KS) {
      uint64_t ko = (uint64_t)(s + 1) * 64u;
      tdm_load_tile(aBase + ko, lds_addr_of(&sA[(s + 1) & 1][0]), 128u, (uint32_t)K,
                    (uint32_t)rowsv);
      tdm_load_tile(gBase + ko, lds_addr_of(&sBg[(s + 1) & 1][0]), 128u, (uint32_t)K,
                    128u);
      tdm_load_tile(uBase + ko, lds_addr_of(&sBu[(s + 1) & 1][0]), 128u, (uint32_t)K,
                    128u);
    }
    if (s == 0)            __builtin_amdgcn_s_wait_tensorcnt(7);
    else if ((s + 1) < KS) __builtin_amdgcn_s_wait_tensorcnt(3);
    else                   __builtin_amdgcn_s_wait_tensorcnt(0);
    __syncthreads();
    v16bf af[2];
#pragma unroll
    for (int i = 0; i < 2; ++i)
      af[i] = load_frag(sA[buf], 32 * wm + 16 * i + (lane & 15), lane);
#pragma unroll
    for (int j = 0; j < 4; ++j) {
      v16bf bg = load_frag(sBg[buf], 64 * wn + 16 * j + (lane & 15), lane);
      accg[0][j] = wmma_bf16(af[0], bg, accg[0][j]);
      accg[1][j] = wmma_bf16(af[1], bg, accg[1][j]);
      v16bf bu = load_frag(sBu[buf], 64 * wn + 16 * j + (lane & 15), lane);
      accu[0][j] = wmma_bf16(af[0], bu, accu[0][j]);
      accu[1][j] = wmma_bf16(af[1], bu, accu[1][j]);
    }
    __syncthreads();
  }

  // LoRA tail: acc += zcat(128x32, adapter-masked) @ BcatT(32x128).
  __builtin_amdgcn_s_wait_tensorcnt(0);
  __syncthreads();
  {
    v16bf zg[2], zu[2];
#pragma unroll
    for (int i = 0; i < 2; ++i) {
      zg[i] = load_frag(sZg, 32 * wm + 16 * i + (lane & 15), lane);
      zu[i] = load_frag(sZu, 32 * wm + 16 * i + (lane & 15), lane);
    }
#pragma unroll
    for (int j = 0; j < 4; ++j) {
      v16bf cg = load_frag(sCg, 64 * wn + 16 * j + (lane & 15), lane);
      accg[0][j] = wmma_bf16(zg[0], cg, accg[0][j]);
      accg[1][j] = wmma_bf16(zg[1], cg, accg[1][j]);
      v16bf cu = load_frag(sCu, 64 * wn + 16 * j + (lane & 15), lane);
      accu[0][j] = wmma_bf16(zu[0], cu, accu[0][j]);
      accu[1][j] = wmma_bf16(zu[1], cu, accu[1][j]);
    }
  }

  // Epilogue: h = silu(gate)*up -> bf16.
  const int col_lo = lane & 15;
#pragma unroll
  for (int i = 0; i < 2; ++i) {
#pragma unroll
    for (int v = 0; v < 8; ++v) {
      int m = ((lane >> 4) << 3) + v;
      int lr = 32 * wm + 16 * i + m;
      if (lr >= rowsv) continue;
      size_t gr = (size_t)(row0 + lr);
#pragma unroll
      for (int j = 0; j < 4; ++j) {
        int col = n0 + 64 * wn + 16 * j + col_lo;
        float g = accg[i][j][v];
        float u = accu[i][j][v];
        float sg = g / (1.0f + __expf(-g));  // silu(gate)
        hs[gr * (size_t)Ntot + col] = tobf(sg * u);
      }
    }
  }
}

// ---------------------------------------------------------------------------
// Down-proj grouped GEMM, same 4x2 wave tiling, single projection.
// Epilogue: atomically add routing_weight * value into out[token][col].
// ---------------------------------------------------------------------------
__global__ __launch_bounds__(256)
void k_gemm_down(const __bf16* __restrict__ Asrt,
                 const __bf16* __restrict__ Wdt,
                 const __bf16* __restrict__ zcatD,
                 const __bf16* __restrict__ BcatDt,
                 const unsigned int* __restrict__ mt_e,
                 const unsigned int* __restrict__ mt_row,
                 const unsigned int* __restrict__ mt_rows,
                 const unsigned int* __restrict__ n_mtiles,
                 const unsigned int* __restrict__ sorted_tok,
                 const float* __restrict__ sorted_rw,
                 float* __restrict__ out) {
  if (blockIdx.x >= n_mtiles[0]) return;
  const int K = I_, Ntot = H_;
  const int e     = (int)mt_e[blockIdx.x];
  const int row0  = (int)mt_row[blockIdx.x];
  const int rowsv = (int)mt_rows[blockIdx.x];
  const int n0    = blockIdx.y * 128;
  const int tid = threadIdx.x, lane = tid & 31;
  const int wm = (tid >> 5) & 3;
  const int wn = tid >> 7;

  __shared__ __attribute__((aligned(16))) __bf16 sA[2][128 * LDSP];
  __shared__ __attribute__((aligned(16))) __bf16 sB[2][128 * LDSP];
  __shared__ __attribute__((aligned(16))) __bf16 sZ[128 * LDSP];
  __shared__ __attribute__((aligned(16))) __bf16 sC[128 * LDSP];

  const v8f vz = {0.f, 0.f, 0.f, 0.f, 0.f, 0.f, 0.f, 0.f};
  v8f acc[2][4];
#pragma unroll
  for (int i = 0; i < 2; ++i)
#pragma unroll
    for (int j = 0; j < 4; ++j) acc[i][j] = vz;

  const int KS = K >> 5;
  const uint64_t aBase = (uint64_t)(uintptr_t)Asrt + ((uint64_t)row0 * K) * 2u;
  const uint64_t wBase =
      (uint64_t)(uintptr_t)Wdt + (((uint64_t)e * Ntot + n0) * (uint64_t)K) * 2u;

  if (tid < 32) {
    tdm_load_tile(aBase, lds_addr_of(&sA[0][0]), 128u, (uint32_t)K, (uint32_t)rowsv);
    tdm_load_tile(wBase, lds_addr_of(&sB[0][0]), 128u, (uint32_t)K, 128u);
    tdm_load_tile((uint64_t)(uintptr_t)zcatD + ((uint64_t)row0 * 32u) * 2u,
                  lds_addr_of(&sZ[0]), 128u, 32u, (uint32_t)rowsv);
    tdm_load_tile((uint64_t)(uintptr_t)BcatDt + (((uint64_t)e * Ntot + n0) * 32u) * 2u,
                  lds_addr_of(&sC[0]), 128u, 32u, 128u);
  }
  for (int s = 0; s < KS; ++s) {
    const int buf = s & 1;
    if (tid < 32 && (s + 1) < KS) {
      uint64_t ko = (uint64_t)(s + 1) * 64u;
      tdm_load_tile(aBase + ko, lds_addr_of(&sA[(s + 1) & 1][0]), 128u, (uint32_t)K,
                    (uint32_t)rowsv);
      tdm_load_tile(wBase + ko, lds_addr_of(&sB[(s + 1) & 1][0]), 128u, (uint32_t)K,
                    128u);
    }
    if (s == 0)            __builtin_amdgcn_s_wait_tensorcnt(4);
    else if ((s + 1) < KS) __builtin_amdgcn_s_wait_tensorcnt(2);
    else                   __builtin_amdgcn_s_wait_tensorcnt(0);
    __syncthreads();
    v16bf af[2];
#pragma unroll
    for (int i = 0; i < 2; ++i)
      af[i] = load_frag(sA[buf], 32 * wm + 16 * i + (lane & 15), lane);
#pragma unroll
    for (int j = 0; j < 4; ++j) {
      v16bf bfg = load_frag(sB[buf], 64 * wn + 16 * j + (lane & 15), lane);
      acc[0][j] = wmma_bf16(af[0], bfg, acc[0][j]);
      acc[1][j] = wmma_bf16(af[1], bfg, acc[1][j]);
    }
    __syncthreads();
  }

  __builtin_amdgcn_s_wait_tensorcnt(0);
  __syncthreads();
  {
    v16bf zf[2];
#pragma unroll
    for (int i = 0; i < 2; ++i)
      zf[i] = load_frag(sZ, 32 * wm + 16 * i + (lane & 15), lane);
#pragma unroll
    for (int j = 0; j < 4; ++j) {
      v16bf cf = load_frag(sC, 64 * wn + 16 * j + (lane & 15), lane);
      acc[0][j] = wmma_bf16(zf[0], cf, acc[0][j]);
      acc[1][j] = wmma_bf16(zf[1], cf, acc[1][j]);
    }
  }

  const int col_lo = lane & 15;
#pragma unroll
  for (int i = 0; i < 2; ++i) {
#pragma unroll
    for (int v = 0; v < 8; ++v) {
      int m = ((lane >> 4) << 3) + v;
      int lr = 32 * wm + 16 * i + m;
      if (lr >= rowsv) continue;
      size_t gr = (size_t)(row0 + lr);
      int tok = (int)sorted_tok[gr];
      float rwv = sorted_rw[gr];
#pragma unroll
      for (int j = 0; j < 4; ++j) {
        int col = n0 + 64 * wn + 16 * j + col_lo;
        atomicAdd(&out[(size_t)tok * H_ + col], rwv * acc[i][j][v]);
      }
    }
  }
}

// ---------------------------------------------------------------------------
extern "C" void kernel_launch(void* const* d_in, const int* in_sizes, int n_in,
                              void* d_out, int out_size, void* d_ws, size_t ws_size,
                              hipStream_t stream) {
  (void)in_sizes; (void)n_in; (void)out_size; (void)ws_size;
  const float* x  = (const float*)d_in[0];
  const int* sel  = (const int*)d_in[1];
  const float* rw = (const float*)d_in[2];
  const int* adp  = (const int*)d_in[3];
  const float* Wg = (const float*)d_in[4];
  const float* Ag = (const float*)d_in[5];
  const float* Bg = (const float*)d_in[6];
  const float* Wu = (const float*)d_in[7];
  const float* Au = (const float*)d_in[8];
  const float* Bu = (const float*)d_in[9];
  const float* Wd = (const float*)d_in[10];
  const float* Ad = (const float*)d_in[11];
  const float* Bd = (const float*)d_in[12];
  float* out = (float*)d_out;

  char* p = (char*)d_ws;
  auto alloc = [&](size_t bytes) -> char* {
    char* r = p;
    p += (bytes + 255) & ~(size_t)255;
    return r;
  };
  unsigned int* counts  = (unsigned int*)alloc(32 * 4);
  unsigned int* offsets = (unsigned int*)alloc(33 * 4);
  unsigned int* cursors = (unsigned int*)alloc(32 * 4);
  unsigned int* nmt     = (unsigned int*)alloc(4);
  unsigned int* mt_e    = (unsigned int*)alloc(512 * 4);
  unsigned int* mt_row  = (unsigned int*)alloc(512 * 4);
  unsigned int* mt_rows = (unsigned int*)alloc(512 * 4);
  unsigned int* s_tok   = (unsigned int*)alloc((size_t)TE_ * 4);
  unsigned int* s_adp   = (unsigned int*)alloc((size_t)TE_ * 4);
  float*        s_rw    = (float*)alloc((size_t)TE_ * 4);
  __bf16* xs     = (__bf16*)alloc((size_t)TE_ * H_ * 2);
  __bf16* hs     = (__bf16*)alloc((size_t)TE_ * I_ * 2);
  __bf16* Wgt    = (__bf16*)alloc((size_t)E_ * I_ * H_ * 2);
  __bf16* Wut    = (__bf16*)alloc((size_t)E_ * I_ * H_ * 2);
  __bf16* Wdt    = (__bf16*)alloc((size_t)E_ * H_ * I_ * 2);
  __bf16* AcatG  = (__bf16*)alloc((size_t)E_ * 32 * H_ * 2);
  __bf16* AcatU  = (__bf16*)alloc((size_t)E_ * 32 * H_ * 2);
  __bf16* AcatD  = (__bf16*)alloc((size_t)E_ * 32 * I_ * 2);
  __bf16* BcatGt = (__bf16*)alloc((size_t)E_ * I_ * 32 * 2);
  __bf16* BcatUt = (__bf16*)alloc((size_t)E_ * I_ * 32 * 2);
  __bf16* BcatDt = (__bf16*)alloc((size_t)E_ * H_ * 32 * 2);
  __bf16* zcatG  = (__bf16*)alloc((size_t)TE_ * 32 * 2);
  __bf16* zcatU  = (__bf16*)alloc((size_t)TE_ * 32 * 2);
  __bf16* zcatD  = (__bf16*)alloc((size_t)TE_ * 32 * 2);

  // --- routing ---
  k_zero_f32<<<(T_ * H_ + 255) / 256, 256, 0, stream>>>(out, T_ * H_);
  k_zero_u32<<<1, 64, 0, stream>>>(counts, 32);
  k_hist<<<TE_ / 256, 256, 0, stream>>>(sel, counts);
  k_scan_tiles<<<1, 32, 0, stream>>>(counts, offsets, cursors, mt_e, mt_row, mt_rows, nmt);
  k_scatter<<<TE_ / 256, 256, 0, stream>>>(sel, adp, rw, cursors, s_tok, s_adp, s_rw);
  k_gather_x<<<TE_, 256, 0, stream>>>(x, s_tok, xs);

  // --- precision/layout prep: fp32 -> bf16, weights to N-major ---
  k_cvt_t<<<E_ * (H_ / 32) * (I_ / 32), 256, 0, stream>>>(Wg, Wgt, H_, I_);
  k_cvt_t<<<E_ * (H_ / 32) * (I_ / 32), 256, 0, stream>>>(Wu, Wut, H_, I_);
  k_cvt_t<<<E_ * (I_ / 32) * (H_ / 32), 256, 0, stream>>>(Wd, Wdt, I_, H_);
  k_cat_A<<<(E_ * 32 * H_) / 256, 256, 0, stream>>>(Ag, AcatG, H_);
  k_cat_A<<<(E_ * 32 * H_) / 256, 256, 0, stream>>>(Au, AcatU, H_);
  k_cat_A<<<(E_ * 32 * I_) / 256, 256, 0, stream>>>(Ad, AcatD, I_);
  k_cat_B<<<(E_ * I_ * 32) / 256, 256, 0, stream>>>(Bg, BcatGt, I_);
  k_cat_B<<<(E_ * I_ * 32) / 256, 256, 0, stream>>>(Bu, BcatUt, I_);
  k_cat_B<<<(E_ * H_ * 32) / 256, 256, 0, stream>>>(Bd, BcatDt, H_);

  // --- LoRA z for gate/up ---
  k_gemm_z<<<dim3(MAXMT, 1), 256, 0, stream>>>(xs, H_, AcatG, mt_e, mt_row, mt_rows,
                                               nmt, s_adp, zcatG);
  k_gemm_z<<<dim3(MAXMT, 1), 256, 0, stream>>>(xs, H_, AcatU, mt_e, mt_row, mt_rows,
                                               nmt, s_adp, zcatU);
  // --- fused gate+up -> h = silu(gate)*up (bf16) ---
  k_gemm_gateup<<<dim3(MAXMT, I_ / 128), 256, 0, stream>>>(
      xs, Wgt, Wut, zcatG, zcatU, BcatGt, BcatUt, mt_e, mt_row, mt_rows, nmt, hs);
  // --- LoRA z for down, then down-proj with routed atomic accumulate ---
  k_gemm_z<<<dim3(MAXMT, 1), 256, 0, stream>>>(hs, I_, AcatD, mt_e, mt_row, mt_rows,
                                               nmt, s_adp, zcatD);
  k_gemm_down<<<dim3(MAXMT, H_ / 128), 256, 0, stream>>>(
      hs, Wdt, zcatD, BcatDt, mt_e, mt_row, mt_rows, nmt, s_tok, s_rw, out);
}
